// GraphAggregator_21440476742361
// MI455X (gfx1250) — matrix operand order repaired and checked by reading, not compiled, not run on hardware
//
#include <hip/hip_runtime.h>
#include <hip/hip_bf16.h>

typedef __bf16 bf16;
typedef __attribute__((ext_vector_type(16))) __bf16 v16bf;
typedef __attribute__((ext_vector_type(8)))  __bf16 v8bf;
typedef __attribute__((ext_vector_type(4)))  __bf16 v4bf;
typedef __attribute__((ext_vector_type(8)))  float  v8f;
typedef __attribute__((ext_vector_type(4)))  float  v4f;

#define DIN   256
#define HID   512
#define DOUT  128
#define WAVES 8
#define RPB   (WAVES * 16)          // 128 rows per block
#define LSTR  520                   // LDS act row stride (bf16), 512 + 8 pad

// --- bf16 fragment pool (elements) --------------------------------------
#define W1_KT 8
#define W2_KT 16
#define W3_KT 16
#define W1_ELEMS (W1_KT * 32 * 512)
#define W2_ELEMS (W2_KT * 32 * 512)
#define W3_ELEMS (W3_KT * 8  * 512)
#define W1_OFF 0
#define W2_OFF (W1_ELEMS)
#define W3_OFF (W1_ELEMS + W2_ELEMS)
#define TOT_ELEMS (W1_ELEMS + W2_ELEMS + W3_ELEMS)

// --- LDS layout (bf16 elements) ------------------------------------------
#define ACT_ELEMS   (WAVES * 2 * 16 * LSTR)   // 133120 elems (260 KB)
#define KC          4                         // k-tiles staged per panel
#define PANEL_ELEMS (4 * KC * 512)            // 8192 elems = 16 KB
#define PAN0_OFF    (ACT_ELEMS)
#define PAN1_OFF    (ACT_ELEMS + PANEL_ELEMS)
#define LDS_TOTAL_BYTES ((ACT_ELEMS + 2 * PANEL_ELEMS) * 2)   // 299008 B

// ---------------------------------------------------------------------------
__global__ void zero_out_kernel(float* __restrict__ out, int n) {
    int t = blockIdx.x * 256 + threadIdx.x;
    if (t < n) out[t] = 0.0f;
}

// ---------------------------------------------------------------------------
// Repack f32 row-major weights into bf16 WMMA B-fragment layout.
// Fragment f = nt*Ktiles + kt: 32 lanes x 16 bf16, lane = (khalf<<4)|n,
// values K = kt*32 + khalf*16 + j (j=0..15), column nt*16 + n.
// ---------------------------------------------------------------------------
__global__ void prep_weights_kernel(const float* __restrict__ W1,
                                    const float* __restrict__ W2,
                                    const float* __restrict__ W3,
                                    bf16* __restrict__ ws) {
    int t = blockIdx.x * 256 + threadIdx.x;
    if (t >= TOT_ELEMS) return;
    const float* W; int Ktiles, Nn, off, u;
    if (t < W2_OFF)      { W = W1; Ktiles = 8;  Nn = 512; off = W1_OFF; u = t - W1_OFF; }
    else if (t < W3_OFF) { W = W2; Ktiles = 16; Nn = 512; off = W2_OFF; u = t - W2_OFF; }
    else                 { W = W3; Ktiles = 16; Nn = 128; off = W3_OFF; u = t - W3_OFF; }
    int j    = u & 15;
    int lane = (u >> 4) & 31;
    int f    = u >> 9;
    int kt   = f % Ktiles;
    int nt   = f / Ktiles;
    int k    = kt * 32 + (lane >> 4) * 16 + j;
    int n    = nt * 16 + (lane & 15);
    ws[off + u] = (bf16)W[(size_t)k * Nn + n];
}

// ---------------------------------------------------------------------------
// Tensor Data Mover staging of one 16 KB weight panel (4 rows x 4 KB,
// row stride KT*512 elems) into LDS.  D# built per CDNA5 ISA.
// This toolchain exposes the 6-arg builtin:
//   (uint32x4 g0, int32x8 g1, int32x4 g2, int32x4 g3, int32x8, i32 cpol)
// ---------------------------------------------------------------------------
#if __has_builtin(__builtin_amdgcn_tensor_load_to_lds)
#define HAS_TDM 1
typedef unsigned int u32x4 __attribute__((ext_vector_type(4)));
typedef int          i32x8 __attribute__((ext_vector_type(8)));
typedef int          i32x4 __attribute__((ext_vector_type(4)));

__device__ __forceinline__ void tdm_stage(const bf16* gsrc, unsigned ldsByteOff,
                                          unsigned rowStrideElems) {
    unsigned long long ga = (unsigned long long)(uintptr_t)gsrc;
    u32x4 g0;
    g0[0] = 1u;                                                 // count=1
    g0[1] = __builtin_amdgcn_groupstaticsize() + ldsByteOff;    // lds_addr
    g0[2] = (unsigned)ga;                                       // global_addr lo
    g0[3] = ((unsigned)(ga >> 32) & 0x01FFFFFFu) | (2u << 30);  // hi | type=2
    i32x8 g1;
    g1[0] = 1 << 16;                           // data_size = 1 (2 bytes)
    g1[1] = 0;                                 // tensor_dim0 = 1<<30 (low16=0)
    g1[2] = 0x4000;                            // tensor_dim0 hi | tensor_dim1 lo
    g1[3] = (int)((2048u << 16) | 0x4000u);    // tile_dim0=2048 | tensor_dim1 hi
    g1[4] = 4;                                 // tile_dim1=4, tile_dim2=0
    g1[5] = (int)rowStrideElems;               // tensor_dim0_stride (lo 32)
    g1[6] = 0;
    g1[7] = 0;
    i32x4 z4 = {};
    i32x8 z8 = {};
    __builtin_amdgcn_tensor_load_to_lds(g0, g1, z4, z4, z8, 0);
}
#else
#define HAS_TDM 0
#endif

// ---------------------------------------------------------------------------
// One MLP layer: acts (LDS bf16 [16][LSTR] per wave) x staged B panels.
// Panels double-buffered, staged by TDM (wave 0), consumed by all 8 waves.
// ---------------------------------------------------------------------------
template<int KT, int NT, bool SILU>
__device__ __forceinline__ void mlp_layer(const bf16* __restrict__ in,
                                          bf16* __restrict__ outl,
                                          const bf16* __restrict__ frag,
                                          const float* __restrict__ bias,
                                          bf16* __restrict__ smemBase,
                                          float* __restrict__ out,
                                          const long long* gid, const int* gval,
                                          int lane, int wave, int tid) {
    const int mrow = lane & 15;
    const int half = lane >> 4;
    constexpr int KSTG   = KT / KC;            // k-chunks per ng group
    constexpr int NSTAGE = (NT / 4) * KSTG;
    bf16* pan0 = smemBase + PAN0_OFF;
    bf16* pan1 = smemBase + PAN1_OFF;

    auto issue = [&](int s) {
        int ng = (s / KSTG) * 4;
        int kc = (s % KSTG) * KC;
        const bf16* src = frag + ((size_t)ng * KT + kc) * 512;
#if HAS_TDM
        if (wave == 0)
            tdm_stage(src, (unsigned)((PAN0_OFF + (s & 1) * PANEL_ELEMS) * 2),
                      (unsigned)(KT * 512));
#else
        bf16* dst = (s & 1) ? pan1 : pan0;
        for (int i = tid; i < PANEL_ELEMS / 8; i += WAVES * 32) {
            int r = i >> 8;                    // 256 16B-chunks per 4KB row
            int c = i & 255;
            *(v8bf*)(dst + i * 8) =
                *(const v8bf*)(src + (size_t)r * KT * 512 + c * 8);
        }
#endif
    };

    issue(0);
#if HAS_TDM
    if (wave == 0) __builtin_amdgcn_s_wait_tensorcnt(0);
#endif
    __syncthreads();

    const v8f vzero = {};
    v8f acc[4];
#pragma unroll 2
    for (int s = 0; s < NSTAGE; ++s) {
        if ((s % KSTG) == 0) { acc[0] = vzero; acc[1] = vzero; acc[2] = vzero; acc[3] = vzero; }
        if (s + 1 < NSTAGE) issue(s + 1);      // prefetch next panel (other buffer)
        const bf16* p = (s & 1) ? pan1 : pan0;
        const int kc = (s % KSTG) * KC;
#pragma unroll
        for (int k2 = 0; k2 < KC; ++k2) {
            union { v16bf v; v8bf q[2]; } a;
            const bf16* ap = in + mrow * LSTR + (kc + k2) * 32 + half * 16;
            a.q[0] = *(const v8bf*)(ap);
            a.q[1] = *(const v8bf*)(ap + 8);
#pragma unroll
            for (int t = 0; t < 4; ++t) {
                v16bf bmat = *(const v16bf*)(p + (t * KC + k2) * 512 + lane * 16);
                acc[t] = __builtin_amdgcn_wmma_f32_16x16x32_bf16(
                    false, a.v, false, bmat, (short)0, acc[t], false, false);
            }
        }
        if ((s % KSTG) == KSTG - 1) {          // epilogue for this ng group
            int ng = (s / KSTG) * 4;
#pragma unroll
            for (int t = 0; t < 4; ++t) {
                float bv = bias[(ng + t) * 16 + mrow];
#pragma unroll
                for (int i = 0; i < 8; ++i) {
                    float c = acc[t][i] + bv;
                    if constexpr (!SILU) {
                        c = fmaxf(c, 0.0f);
                        outl[(i + 8 * half) * LSTR + (ng + t) * 16 + mrow] = (bf16)c;
                    } else {
                        float g = c / (1.0f + __expf(-c));     // SiLU
                        if (gval[i])
                            unsafeAtomicAdd(out + (size_t)gid[i] * DOUT +
                                            (ng + t) * 16 + mrow, g);
                    }
                }
            }
        }
#if HAS_TDM
        if (wave == 0 && s + 1 < NSTAGE) __builtin_amdgcn_s_wait_tensorcnt(0);
#endif
        __syncthreads();
    }
}

// ---------------------------------------------------------------------------
__global__ __launch_bounds__(WAVES * 32) void mlp_silu_segsum_kernel(
    const float* __restrict__ X,
    const long long* __restrict__ gidx,
    const bf16* __restrict__ wf,
    const float* __restrict__ b1, const float* __restrict__ b2,
    const float* __restrict__ b3,
    float* __restrict__ out, int nRows)
{
    extern __shared__ bf16 smem[];
    const int tid  = threadIdx.x;
    const int lane = tid & 31;
    const int wave = tid >> 5;
    bf16* bufA = smem + (size_t)wave * (2 * 16 * LSTR);
    bf16* bufB = bufA + 16 * LSTR;
    const int rowBase = blockIdx.x * RPB + wave * 16;

    // ---- stage X tile [16][256] f32 -> bf16 (coalesced float4 loads)
#pragma unroll 4
    for (int i = 0; i < 32; ++i) {
        int idx = i * 32 + lane;
        int r  = idx >> 6;
        int c4 = idx & 63;
        int gr = rowBase + r;
        if (gr >= nRows) gr = nRows - 1;        // clamp; masked at atomics
        v4f v = *(const v4f*)(X + (size_t)gr * DIN + (size_t)c4 * 4);
        v4bf pk;
        pk[0] = (bf16)v[0]; pk[1] = (bf16)v[1];
        pk[2] = (bf16)v[2]; pk[3] = (bf16)v[3];
        *(v4bf*)(bufA + r * LSTR + c4 * 4) = pk;
    }

    const int half = lane >> 4;
    long long gid[8];
    int gval[8];
#pragma unroll
    for (int i = 0; i < 8; ++i) {
        int gr = rowBase + i + 8 * half;
        gval[i] = (gr < nRows);
        gid[i]  = gidx[gval[i] ? gr : (nRows - 1)];
    }

    // L1: [16x256]@[256x512]+b1 relu   (bufA -> bufB)
    mlp_layer<W1_KT, 32, false>(bufA, bufB, wf + W1_OFF, b1, smem,
                                nullptr, gid, gval, lane, wave, tid);
    // L2: [16x512]@[512x512]+b2 relu   (bufB -> bufA)
    mlp_layer<W2_KT, 32, false>(bufB, bufA, wf + W2_OFF, b2, smem,
                                nullptr, gid, gval, lane, wave, tid);
    // L3: [16x512]@[512x128]+b3, SiLU, atomic segment-sum
    mlp_layer<W3_KT, 8,  true >(bufA, nullptr, wf + W3_OFF, b3, smem,
                                out, gid, gval, lane, wave, tid);
}

// ---------------------------------------------------------------------------
extern "C" void kernel_launch(void* const* d_in, const int* in_sizes, int n_in,
                              void* d_out, int out_size, void* d_ws, size_t ws_size,
                              hipStream_t stream) {
    const float*     X    = (const float*)d_in[0];
    const long long* gidx = (const long long*)d_in[1];   // int64 graph ids
    const float*     W1   = (const float*)d_in[2];
    const float*     b1   = (const float*)d_in[3];
    const float*     W2   = (const float*)d_in[4];
    const float*     b2   = (const float*)d_in[5];
    const float*     W3   = (const float*)d_in[6];
    const float*     b3   = (const float*)d_in[7];
    float* out = (float*)d_out;
    bf16*  wf  = (bf16*)d_ws;                  // 896 KB bf16 fragment pool
    int nRows = in_sizes[0] / DIN;

    (void)hipFuncSetAttribute((const void*)mlp_silu_segsum_kernel,
                              hipFuncAttributeMaxDynamicSharedMemorySize,
                              (int)LDS_TOTAL_BYTES);

    zero_out_kernel<<<(out_size + 255) / 256, 256, 0, stream>>>(out, out_size);
    prep_weights_kernel<<<(TOT_ELEMS + 255) / 256, 256, 0, stream>>>(W1, W2, W3, wf);
    int blocks = (nRows + RPB - 1) / RPB;
    mlp_silu_segsum_kernel<<<blocks, WAVES * 32, (size_t)LDS_TOTAL_BYTES, stream>>>(
        X, gidx, wf, b1, b2, b3, out, nRows);
}
